// Convolution_75960791597065
// MI455X (gfx1250) — matrix operand order repaired and checked by reading, not compiled, not run on hardware
//
#include <hip/hip_runtime.h>
#include <math.h>

#define Nn 50000
#define Ee 500000

#define INV0   0.125f                 // 1/sqrt(64)
#define INV1   0.17677669529663687f   // 1/sqrt(32)
#define INV16  0.25f                  // 1/sqrt(16)
#define INVH   0.125f                 // 1/sqrt(64)
#define INV3   0.5773502691896258f    // 1/sqrt(3)
#define INV_NB 0.31622776601683794f   // 1/sqrt(10)
#define INV_MID 0.10206207261596575f  // 1/sqrt(96)
#define C_Sc   0.3826834323650898f    // sin(pi/8)
#define C_Xc   0.9238795325112867f    // cos(pi/8)

typedef __attribute__((ext_vector_type(16))) _Float16 v16h;
typedef __attribute__((ext_vector_type(8)))  float    v8f;
typedef __attribute__((ext_vector_type(2)))  float    v2f;

// ---------------------------------------------------------------------------
// zero workspace region
// ---------------------------------------------------------------------------
__global__ void zero_kernel(float* __restrict__ p, long long n) {
    long long i = (long long)blockIdx.x * blockDim.x + threadIdx.x;
    long long stride = (long long)gridDim.x * blockDim.x;
    for (; i < n; i += stride) p[i] = 0.f;
}

// ---------------------------------------------------------------------------
// node input linears via f32 WMMA (16x16x4, full precision):
//   out[:, :] = C_S * s   (self term), workspace y0/y1 (message source).
// One wave per 16-node tile; weight fragments pre-staged in LDS per block.
// Fragment layout (ISA 7.12.2, 32-bit A 16x4): lane = M (low/high half),
// per lane a pair (X[m][k], X[m][k+1]) with k = 4*kt + 2*(lane>>4).
// B symmetric: pair (W[k][col], W[k+1][col]), lane = column.
// ---------------------------------------------------------------------------
__global__ __launch_bounds__(256) void node_in_kernel(
    const float* __restrict__ xin,  const float* __restrict__ attr,
    const float* __restrict__ Wsc0, const float* __restrict__ Wsc1,
    const float* __restrict__ Wl10, const float* __restrict__ Wl11,
    float* __restrict__ out, float* __restrict__ y0, float* __restrict__ y1)
{
    __shared__ __align__(16) float Fsc0[4 * 16 * 32 * 2];  // 16 KB
    __shared__ __align__(16) float Fl10[4 * 16 * 32 * 2];  // 16 KB
    __shared__ __align__(16) float Fsc1[2 * 8 * 32 * 2];   //  4 KB
    __shared__ __align__(16) float Fl11[2 * 8 * 32 * 2];   //  4 KB

    const int tid  = threadIdx.x;
    const int w    = tid >> 5;
    const int lane = tid & 31;
    const int lo   = lane & 15;
    const int kh   = lane >> 4;

    // stage 64x64 weight fragments (pair-contiguous per lane)
    for (int idx = tid; idx < 4 * 16 * 32; idx += 256) {
        int ct = idx >> 9, kt = (idx >> 5) & 15, ln = idx & 31;
        int k   = 4 * kt + 2 * (ln >> 4);
        int col = 16 * ct + (ln & 15);
        Fsc0[idx * 2 + 0] = Wsc0[k * 64 + col];
        Fsc0[idx * 2 + 1] = Wsc0[(k + 1) * 64 + col];
        Fl10[idx * 2 + 0] = Wl10[k * 64 + col];
        Fl10[idx * 2 + 1] = Wl10[(k + 1) * 64 + col];
    }
    // stage 32x32 weight fragments
    for (int idx = tid; idx < 2 * 8 * 32; idx += 256) {
        int ct = idx >> 8, kt = (idx >> 5) & 7, ln = idx & 31;
        int k   = 4 * kt + 2 * (ln >> 4);
        int col = 16 * ct + (ln & 15);
        Fsc1[idx * 2 + 0] = Wsc1[k * 32 + col];
        Fsc1[idx * 2 + 1] = Wsc1[(k + 1) * 32 + col];
        Fl11[idx * 2 + 0] = Wl11[k * 32 + col];
        Fl11[idx * 2 + 1] = Wl11[(k + 1) * 32 + col];
    }
    __syncthreads();

    const int NT = Nn / 16;          // 3125 node tiles
    const int CH = (NT + 7) / 8;     // 8 waves/block
    for (int ch = blockIdx.x; ch < CH; ch += gridDim.x) {
        const int tile = ch * 8 + w;
        if (tile >= NT) continue;     // wave-uniform, no barriers below
        const int nbase = tile * 16;

        float av[8];                  // attrs of the 8 rows this lane owns
        #pragma unroll
        for (int j = 0; j < 8; ++j) av[j] = attr[nbase + j + 8 * kh];

        const float* xp = xin + (size_t)(nbase + lo) * 160;

        // ---- scalar channels: x0(16x64) @ {Wsc0,Wl10} ----
        v8f sa[4], ya[4];
        #pragma unroll
        for (int ct = 0; ct < 4; ++ct) { sa[ct] = (v8f){}; ya[ct] = (v8f){}; }
        for (int kt = 0; kt < 16; ++kt) {
            v2f a = *(const v2f*)(xp + 4 * kt + 2 * kh);
            #pragma unroll
            for (int ct = 0; ct < 4; ++ct) {
                v2f bs = *(const v2f*)&Fsc0[((ct * 16 + kt) * 32 + lane) * 2];
                sa[ct] = __builtin_amdgcn_wmma_f32_16x16x4_f32(
                             false, a, false, bs, (short)0, sa[ct], false, false);
                v2f by = *(const v2f*)&Fl10[((ct * 16 + kt) * 32 + lane) * 2];
                ya[ct] = __builtin_amdgcn_wmma_f32_16x16x4_f32(
                             false, a, false, by, (short)0, ya[ct], false, false);
            }
        }
        #pragma unroll
        for (int ct = 0; ct < 4; ++ct) {
            #pragma unroll
            for (int j = 0; j < 8; ++j) {
                int node = nbase + j + 8 * kh;
                int col  = 16 * ct + lo;
                out[(size_t)node * 160 + col] = C_Sc * (sa[ct][j] * av[j] * INV0);
                y0 [(size_t)node * 64  + col] = ya[ct][j] * av[j] * INV0;
            }
        }

        // ---- vector channels: per component, x1c(16x32) @ {Wsc1,Wl11} ----
        for (int cc = 0; cc < 3; ++cc) {
            v8f s1[2], y1a[2];
            s1[0] = (v8f){}; s1[1] = (v8f){};
            y1a[0] = (v8f){}; y1a[1] = (v8f){};
            for (int kt = 0; kt < 8; ++kt) {
                int k = 4 * kt + 2 * kh;
                v2f a;
                a[0] = xp[64 + 3 * k + cc];
                a[1] = xp[64 + 3 * (k + 1) + cc];
                #pragma unroll
                for (int ct = 0; ct < 2; ++ct) {
                    v2f bs = *(const v2f*)&Fsc1[((ct * 8 + kt) * 32 + lane) * 2];
                    s1[ct] = __builtin_amdgcn_wmma_f32_16x16x4_f32(
                                 false, a, false, bs, (short)0, s1[ct], false, false);
                    v2f by = *(const v2f*)&Fl11[((ct * 8 + kt) * 32 + lane) * 2];
                    y1a[ct] = __builtin_amdgcn_wmma_f32_16x16x4_f32(
                                 false, a, false, by, (short)0, y1a[ct], false, false);
                }
            }
            #pragma unroll
            for (int ct = 0; ct < 2; ++ct) {
                #pragma unroll
                for (int j = 0; j < 8; ++j) {
                    int node = nbase + j + 8 * kh;
                    int v    = 16 * ct + lo;
                    out[(size_t)node * 160 + 64 + 3 * v + cc] =
                        C_Sc * (s1[ct][j] * av[j] * INV1);
                    y1 [(size_t)node * 96 + 3 * v + cc] =
                        y1a[ct][j] * av[j] * INV1;
                }
            }
        }
    }
}

// ---------------------------------------------------------------------------
// fused edge kernel: MLP (WMMA f16) + tensor product + scatter atomics.
// 2 waves per block; each wave owns one 16-edge tile per iteration.
// ---------------------------------------------------------------------------
__global__ __launch_bounds__(64) void edge_kernel(
    const float* __restrict__ ele,  const float* __restrict__ fcW1,
    const float* __restrict__ fcW2, const int* __restrict__ esrc,
    const int* __restrict__ edst,   const float* __restrict__ eattr,
    const float* __restrict__ y0g,  const float* __restrict__ y1g,
    float* __restrict__ z0,         float* __restrict__ z1)
{
    __shared__ __align__(32) _Float16 W1h[4  * 32 * 16];   //  4 KB
    __shared__ __align__(32) _Float16 W2h[24 * 32 * 16];   // 24 KB
    __shared__ __align__(16) _Float16 Hh[2][16][64];       //  4 KB
    __shared__ float Wbuf[2][16][192];                     // 24 KB
    __shared__ int   Es[2][16], Ed[2][16];
    __shared__ float Ea[2][16][4];

    const int tid  = threadIdx.x;
    const int w    = tid >> 5;
    const int lane = tid & 31;
    const int lo   = lane & 15;
    const int kh   = lane >> 4;

    for (int idx = tid; idx < 4 * 32 * 16; idx += 64) {
        int nt = idx >> 9, rem = idx & 511;
        int ln = rem >> 4, i = rem & 15;
        int col = nt * 16 + (ln & 15);
        int k   = (i < 8) ? (ln >> 4) * 8 + i : 16 + (ln >> 4) * 8 + (i - 8);
        W1h[idx] = (k < 16) ? (_Float16)(fcW1[k * 64 + col] * INV16)
                            : (_Float16)0.f;
    }
    for (int idx = tid; idx < 24 * 32 * 16; idx += 64) {
        int frag = idx >> 9, rem = idx & 511;      // frag = nt2*2 + kc
        int ln = rem >> 4, i = rem & 15;
        int nt2 = frag >> 1, kc = frag & 1;
        int col = nt2 * 16 + (ln & 15);
        int k   = kc * 32 + ((i < 8) ? (ln >> 4) * 8 + i : 16 + (ln >> 4) * 8 + (i - 8));
        W2h[idx] = (_Float16)(fcW2[k * 192 + col] * INVH);
    }
    __syncthreads();

    const int TP = (Ee / 16) / 2;   // tile pairs; both waves iterate equally
    for (int t = blockIdx.x; t < TP; t += gridDim.x) {
        const int tile  = t * 2 + w;
        const int ebase = tile * 16;

        // prefetch next iteration's embedding rows into cache
        if (t + (int)gridDim.x < TP) {
            const int nb = ((t + gridDim.x) * 2 + w) * 16;
            __builtin_prefetch(ele + (size_t)(nb + lo) * 16 + kh * 8, 0, 1);
        }

        if (lane < 16) {
            int e = ebase + lane;
            Es[w][lane] = esrc[e];
            Ed[w][lane] = edst[e];
            Ea[w][lane][0] = eattr[(size_t)e * 4 + 0];
            Ea[w][lane][1] = eattr[(size_t)e * 4 + 1];
            Ea[w][lane][2] = eattr[(size_t)e * 4 + 2];
            Ea[w][lane][3] = eattr[(size_t)e * 4 + 3];
        }

        // A fragment: 16 edges x K32 (K>=16 zero)
        v16h afrag;
        {
            const float* p = ele + (size_t)(ebase + lo) * 16 + kh * 8;
            #pragma unroll
            for (int i = 0; i < 8; ++i) afrag[i] = (_Float16)p[i];
            #pragma unroll
            for (int i = 8; i < 16; ++i) afrag[i] = (_Float16)0.f;
        }

        // layer 1: 4 output tiles, silu, stage h row-major in LDS
        #pragma unroll
        for (int nt = 0; nt < 4; ++nt) {
            v16h b = *(const v16h*)&W1h[(nt * 32 + lane) * 16];
            v8f  c = {};
            c = __builtin_amdgcn_wmma_f32_16x16x32_f16(
                    false, afrag, false, b, (short)0, c, false, false);
            #pragma unroll
            for (int j = 0; j < 8; ++j) {
                float x  = c[j];
                float sv = x / (1.f + __expf(-x));       // silu
                Hh[w][j + 8 * kh][nt * 16 + lo] = (_Float16)sv;
            }
        }
        __syncthreads();

        // re-fragment h (16x64) into two 16x32 A fragments
        v16h a2[2];
        #pragma unroll
        for (int kc = 0; kc < 2; ++kc) {
            #pragma unroll
            for (int i = 0; i < 8; ++i)
                a2[kc][i] = Hh[w][lo][kc * 32 + kh * 8 + i];
            #pragma unroll
            for (int i = 8; i < 16; ++i)
                a2[kc][i] = Hh[w][lo][kc * 32 + 16 + kh * 8 + (i - 8)];
        }

        // layer 2: 12 output tiles x 2 K-chunks
        #pragma unroll
        for (int nt2 = 0; nt2 < 12; ++nt2) {
            v8f  c  = {};
            v16h b0 = *(const v16h*)&W2h[((nt2 * 2 + 0) * 32 + lane) * 16];
            c = __builtin_amdgcn_wmma_f32_16x16x32_f16(
                    false, a2[0], false, b0, (short)0, c, false, false);
            v16h b1 = *(const v16h*)&W2h[((nt2 * 2 + 1) * 32 + lane) * 16];
            c = __builtin_amdgcn_wmma_f32_16x16x32_f16(
                    false, a2[1], false, b1, (short)0, c, false, false);
            #pragma unroll
            for (int j = 0; j < 8; ++j)
                Wbuf[w][j + 8 * kh][nt2 * 16 + lo] = c[j];
        }
        __syncthreads();

        // tensor product + scatter; 32 lanes cooperate per edge
        for (int e = 0; e < 16; ++e) {
            const int   src = Es[w][e], dst = Ed[w][e];
            const float e0  = Ea[w][e][0];
            const float e1x = Ea[w][e][1], e1y = Ea[w][e][2], e1z = Ea[w][e][3];
            const float* wrow = &Wbuf[w][e][0];

            #pragma unroll
            for (int r = 0; r < 2; ++r) {
                int   uu  = lane + 32 * r;
                float ys0 = y0g[(size_t)src * 64 + uu];
                float wa  = wrow[uu], wb = wrow[64 + uu];
                atomicAdd(&z0[(size_t)dst * 96 + uu], wa * ys0 * e0 * INV_NB);
                float tb = wb * ys0 * INV_NB;
                float* zp = &z1[(size_t)dst * 288 + uu * 3];
                atomicAdd(zp + 0, tb * e1x);
                atomicAdd(zp + 1, tb * e1y);
                atomicAdd(zp + 2, tb * e1z);
            }
            {
                int u = lane;
                const float* yp = &y1g[(size_t)src * 96 + u * 3];
                float yx = yp[0], yy = yp[1], yz = yp[2];
                float wc = wrow[128 + u], wd = wrow[160 + u];
                float dot = yx * e1x + yy * e1y + yz * e1z;
                atomicAdd(&z0[(size_t)dst * 96 + 64 + u],
                          wd * dot * INV3 * INV_NB);
                float tc = wc * e0 * INV_NB;
                float* zp = &z1[(size_t)dst * 288 + (64 + u) * 3];
                atomicAdd(zp + 0, tc * yx);
                atomicAdd(zp + 1, tc * yy);
                atomicAdd(zp + 2, tc * yz);
            }
        }
        __syncthreads();
    }
}

// ---------------------------------------------------------------------------
// node output linears: out += C_X * a * inv_mid * (z @ W_l2)
// weights LDS-cached once per block; 32 nodes per block.
// ---------------------------------------------------------------------------
__global__ __launch_bounds__(160) void node_out_kernel(
    const float* __restrict__ z0, const float* __restrict__ z1,
    const float* __restrict__ attr,
    const float* __restrict__ Wl20, const float* __restrict__ Wl21,
    float* __restrict__ out)
{
    __shared__ float W0s[96 * 64];   // 24 KB
    __shared__ float W1s[96 * 32];   // 12 KB
    __shared__ float zr0[96];
    __shared__ float zr1[288];
    const int c = threadIdx.x;
    for (int i = c; i < 96 * 64; i += 160) W0s[i] = Wl20[i];
    for (int i = c; i < 96 * 32; i += 160) W1s[i] = Wl21[i];
    __syncthreads();

    const int NPB = 32;
    const int n0 = blockIdx.x * NPB;
    for (int nn = 0; nn < NPB; ++nn) {
        const int n = n0 + nn;
        if (n >= Nn) break;          // uniform across block
        if (c < 96) zr0[c] = z0[(size_t)n * 96 + c];
        for (int i = c; i < 288; i += 160) zr1[i] = z1[(size_t)n * 288 + i];
        __syncthreads();
        const float a = attr[n];
        float dot = 0.f;
        if (c < 64) {
            #pragma unroll 4
            for (int u = 0; u < 96; ++u) dot += zr0[u] * W0s[u * 64 + c];
        } else {
            const int r = c - 64, v = r / 3, ccd = r % 3;
            #pragma unroll 4
            for (int u = 0; u < 96; ++u) dot += zr1[u * 3 + ccd] * W1s[u * 32 + v];
        }
        const size_t o = (size_t)n * 160 + c;
        out[o] = out[o] + C_Xc * a * INV_MID * dot;
        __syncthreads();
    }
}

// ---------------------------------------------------------------------------
extern "C" void kernel_launch(void* const* d_in, const int* in_sizes, int n_in,
                              void* d_out, int out_size, void* d_ws, size_t ws_size,
                              hipStream_t stream) {
    (void)in_sizes; (void)n_in; (void)out_size; (void)ws_size;

    const float* node_input = (const float*)d_in[0];
    const float* node_attr  = (const float*)d_in[1];
    const int*   edge_src   = (const int*)  d_in[2];
    const int*   edge_dst   = (const int*)  d_in[3];
    const float* edge_attr  = (const float*)d_in[4];
    const float* ele        = (const float*)d_in[5];
    const float* Wsc0       = (const float*)d_in[6];
    const float* Wsc1       = (const float*)d_in[7];
    const float* Wl10       = (const float*)d_in[8];
    const float* Wl11       = (const float*)d_in[9];
    const float* fcW1       = (const float*)d_in[10];
    const float* fcW2       = (const float*)d_in[11];
    const float* Wl20       = (const float*)d_in[12];
    const float* Wl21       = (const float*)d_in[13];

    float* out = (float*)d_out;
    float* ws  = (float*)d_ws;
    float* y0  = ws;                       // N*64
    float* y1  = ws + (size_t)Nn * 64;     // N*96
    float* z0  = ws + (size_t)Nn * 160;    // N*96
    float* z1  = ws + (size_t)Nn * 256;    // N*288  (z region contiguous)

    zero_kernel<<<2048, 256, 0, stream>>>(z0, (long long)Nn * 384);

    node_in_kernel<<<391, 256, 0, stream>>>(node_input, node_attr,
                                            Wsc0, Wsc1, Wl10, Wl11,
                                            out, y0, y1);

    edge_kernel<<<1536, 64, 0, stream>>>(ele, fcW1, fcW2,
                                         edge_src, edge_dst, edge_attr,
                                         y0, y1, z0, z1);

    node_out_kernel<<<(Nn + 31) / 32, 160, 0, stream>>>(z0, z1, node_attr,
                                                        Wl20, Wl21, out);
}
